// GPTModel_39797166964898
// MI455X (gfx1250) — matrix operand order repaired and checked by reading, not compiled, not run on hardware
//
#include <hip/hip_runtime.h>

// ---------------------------------------------------------------------------
// GPT forward for MI455X (gfx1250), wave32, WMMA f32_16x16x32_f16 everywhere.
// GEMM: 128x128 block tile, 8 waves, 8 WMMAs/wave/K-step (32x64 per wave).
// ---------------------------------------------------------------------------

typedef _Float16 half_t;
typedef __attribute__((ext_vector_type(16))) _Float16 v16h;
typedef __attribute__((ext_vector_type(8)))  _Float16 v8h;
typedef __attribute__((ext_vector_type(8)))  float    v8f;
typedef __attribute__((ext_vector_type(4)))  float    v4f;

#define VOCAB  32000
#define NLAYER 4
#define DMODEL 512
#define NHEAD  8
#define HSZ    64
#define TBLK   2048
#define BATCH  2
#define MROWS  (BATCH * TBLK)   // 4096
#define DFF    (4 * DMODEL)     // 2048

__device__ __forceinline__ v8f zero8() {
  v8f z = {0.f, 0.f, 0.f, 0.f, 0.f, 0.f, 0.f, 0.f};
  return z;
}

__device__ __forceinline__ v16h comb(v8h lo, v8h hi) {
  return __builtin_shufflevector(lo, hi, 0, 1, 2, 3, 4, 5, 6, 7,
                                         8, 9, 10, 11, 12, 13, 14, 15);
}

// ---------------------------------------------------------------------------
// Embedding: x[b,t,:] = tok_emb[idx[b,t],:] + pos_emb[t,:]
// ---------------------------------------------------------------------------
__global__ void __launch_bounds__(256)
embed_kernel(const int* __restrict__ idx, const float* __restrict__ tok,
             const float* __restrict__ pos, float* __restrict__ x) {
  const int i   = blockIdx.x * 256 + threadIdx.x;   // over MROWS*DMODEL
  const int row = i >> 9;                           // /512
  const int c   = i & (DMODEL - 1);
  const int t   = row & (TBLK - 1);
  x[i] = tok[(size_t)idx[row] * DMODEL + c] + pos[(size_t)t * DMODEL + c];
}

// ---------------------------------------------------------------------------
// LayerNorm: one wave per row of 512; lane holds 16 contiguous floats.
// ---------------------------------------------------------------------------
__global__ void __launch_bounds__(256)
ln_kernel(const float* __restrict__ x, const float* __restrict__ g,
          const float* __restrict__ b, float* __restrict__ out) {
  const int lane = threadIdx.x & 31;
  const int wave = threadIdx.x >> 5;
  const int row  = blockIdx.x * 8 + wave;
  const float* xr = x + (size_t)row * DMODEL;
  v4f v[4];
  float s = 0.f, sq = 0.f;
#pragma unroll
  for (int j = 0; j < 4; ++j) {
    v[j] = *(const v4f*)(xr + lane * 16 + j * 4);
#pragma unroll
    for (int e = 0; e < 4; ++e) { s += v[j][e]; sq += v[j][e] * v[j][e]; }
  }
#pragma unroll
  for (int off = 1; off < 32; off <<= 1) {
    s  += __shfl_xor(s, off, 32);
    sq += __shfl_xor(sq, off, 32);
  }
  const float mean = s * (1.f / DMODEL);
  const float var  = sq * (1.f / DMODEL) - mean * mean;
  const float rs   = rsqrtf(var + 1e-5f);
  float* outr = out + (size_t)row * DMODEL;
#pragma unroll
  for (int j = 0; j < 4; ++j) {
    const int c = lane * 16 + j * 4;
    v4f gg = *(const v4f*)(g + c);
    v4f bb = *(const v4f*)(b + c);
    v4f o;
#pragma unroll
    for (int e = 0; e < 4; ++e) o[e] = (v[j][e] - mean) * rs * gg[e] + bb[e];
    *(v4f*)(outr + c) = o;
  }
}

// ---------------------------------------------------------------------------
// Generic WMMA GEMM: C[M,N] = A[M,K] @ W[K,N] (+bias, + epilogue variant).
// Block = 256 thr (8 waves), 128x128 tile, K-step 32. A and W staged to LDS
// with f32->f16 conversion; W stored n-major so B fragments are 32B reads.
// Each wave owns a 32x64 sub-tile: 2 A frags x 4 B frags -> 8 WMMAs / K-step.
// ---------------------------------------------------------------------------
enum { MODE_OUT = 0, MODE_QK = 1, MODE_V = 2, MODE_RES = 3, MODE_RELU = 4 };

template <typename AT, int MODE>
__global__ void __launch_bounds__(256)
gemm_kernel(const AT* __restrict__ A, const float* __restrict__ W,
            const float* __restrict__ bias, void* __restrict__ outp,
            int M, int N, int K) {
  __shared__ half_t As[128 * 32];   // row-major [128 m][32 k]
  __shared__ half_t Bt[128 * 32];   // n-major   [128 n][32 k]
  (void)M;
  const int tid  = threadIdx.x;
  const int lane = tid & 31;
  const int wave = tid >> 5;
  const int lm = lane & 15, lg = lane >> 4;
  const int nBase = blockIdx.x * 128;
  const int mBase = blockIdx.y * 128;
  const int mt2 = (wave & 3) * 2;        // first of 2 M sub-tiles (16 rows each)
  const int nt4 = (wave >> 2) * 4;       // first of 4 N sub-tiles
  const int arow = tid >> 1, ach = (tid & 1) * 16;  // A staging: 128 rows x 2 chunks of 16
  const int bk = tid >> 3,  bcg = tid & 7;          // W staging: 32 k x 8 col-groups of 16

  v8f acc[2][4];
#pragma unroll
  for (int mi = 0; mi < 2; ++mi)
#pragma unroll
    for (int ni = 0; ni < 4; ++ni) acc[mi][ni] = zero8();

  const AT*    aptr = A + (size_t)(mBase + arow) * K + ach;
  const float* wrow = W + (size_t)bk * N + nBase + bcg * 16;

  for (int k0 = 0; k0 < K; k0 += 32) {
    // ---- stage A tile (f32 or f16 -> f16), 16 elems / thread ----
    {
      v8h a0, a1;
      if (sizeof(AT) == 4) {
        const float* ap = (const float*)aptr + k0;
        v4f f0 = *(const v4f*)ap;
        v4f f1 = *(const v4f*)(ap + 4);
        v4f f2 = *(const v4f*)(ap + 8);
        v4f f3 = *(const v4f*)(ap + 12);
#pragma unroll
        for (int j = 0; j < 4; ++j) {
          a0[j] = (half_t)f0[j]; a0[4 + j] = (half_t)f1[j];
          a1[j] = (half_t)f2[j]; a1[4 + j] = (half_t)f3[j];
        }
      } else {
        const half_t* ap = (const half_t*)aptr + k0;
        a0 = *(const v8h*)ap;
        a1 = *(const v8h*)(ap + 8);
      }
      *(v8h*)&As[arow * 32 + ach]     = a0;
      *(v8h*)&As[arow * 32 + ach + 8] = a1;
    }
    // ---- stage W tile transposed (f32 -> f16), 16 elems / thread ----
    {
      const float* wp = wrow + (size_t)k0 * N;
      v4f w0 = *(const v4f*)wp;
      v4f w1 = *(const v4f*)(wp + 4);
      v4f w2 = *(const v4f*)(wp + 8);
      v4f w3 = *(const v4f*)(wp + 12);
#pragma unroll
      for (int j = 0; j < 4; ++j) {
        Bt[(bcg * 16 + j) * 32 + bk]      = (half_t)w0[j];
        Bt[(bcg * 16 + 4 + j) * 32 + bk]  = (half_t)w1[j];
        Bt[(bcg * 16 + 8 + j) * 32 + bk]  = (half_t)w2[j];
        Bt[(bcg * 16 + 12 + j) * 32 + bk] = (half_t)w3[j];
      }
      if (k0 + 32 < K) {
        __builtin_prefetch((const AT*)aptr + k0 + 32, 0, 1);  // global_prefetch_b8
        __builtin_prefetch(wp + (size_t)32 * N, 0, 1);
      }
    }
    __syncthreads();
    // ---- fragments + 8 WMMAs ----
    v16h a[2];
#pragma unroll
    for (int mi = 0; mi < 2; ++mi) {
      const int r = ((mt2 + mi) * 16 + lm) * 32;
      a[mi] = comb(*(const v8h*)&As[r + 8 * lg], *(const v8h*)&As[r + 16 + 8 * lg]);
    }
#pragma unroll
    for (int ni = 0; ni < 4; ++ni) {
      const int r = ((nt4 + ni) * 16 + lm) * 32 + 8 * lg;
      const v16h b = comb(*(const v8h*)&Bt[r], *(const v8h*)&Bt[r + 16]);
#pragma unroll
      for (int mi = 0; mi < 2; ++mi)
        acc[mi][ni] = __builtin_amdgcn_wmma_f32_16x16x32_f16(false, a[mi], false, b,
                                                             (short)0, acc[mi][ni],
                                                             false, false);
    }
    __syncthreads();
  }

  // ---- epilogue ----
  float*  outf = (float*)outp;
  half_t* outh = (half_t*)outp;
  float bb[4];
#pragma unroll
  for (int ni = 0; ni < 4; ++ni) {
    bb[ni] = 0.f;
    if (MODE == MODE_OUT || MODE == MODE_RES || MODE == MODE_RELU)
      bb[ni] = bias[nBase + (nt4 + ni) * 16 + lm];
  }
#pragma unroll
  for (int mi = 0; mi < 2; ++mi) {
#pragma unroll
    for (int r = 0; r < 8; ++r) {
      const int row = mBase + (mt2 + mi) * 16 + r + 8 * lg;  // C layout: M = r + 8g
#pragma unroll
      for (int ni = 0; ni < 4; ++ni) {
        const int col = nBase + (nt4 + ni) * 16 + lm;
        const float v = acc[mi][ni][r] + bb[ni];
        if (MODE == MODE_OUT) {
          outf[(size_t)row * N + col] = v;
        } else if (MODE == MODE_RES) {
          outf[(size_t)row * N + col] += v;
        } else if (MODE == MODE_RELU) {
          outh[(size_t)row * N + col] = (half_t)fmaxf(v, 0.f);
        } else if (MODE == MODE_QK) {      // f16 [B,H,T,HS]
          const int bi = row >> 11, t = row & (TBLK - 1);
          outh[(((size_t)bi * NHEAD + (col >> 6)) * TBLK + t) * HSZ + (col & 63)] =
              (half_t)v;
        } else {                           // MODE_V: f16 [B,H,HS,T] (pre-transposed)
          const int bi = row >> 11, t = row & (TBLK - 1);
          outh[(((size_t)bi * NHEAD + (col >> 6)) * HSZ + (col & 63)) * TBLK + t] =
              (half_t)v;
        }
      }
    }
  }
}

// ---------------------------------------------------------------------------
// Flash attention: one wave per (b,h,16-row query tile); 32-key steps with
// online softmax. S = Q@K^T (2 tiles x 2 WMMAs), P relayout via 1KB LDS slab,
// O += P@V (4 WMMAs over HS=64). scale = 1/sqrt(D) (faithful to reference).
// ---------------------------------------------------------------------------
__global__ void __launch_bounds__(256)
attn_kernel(const half_t* __restrict__ q, const half_t* __restrict__ k,
            const half_t* __restrict__ vt, float* __restrict__ att) {
  __shared__ half_t Plds[8][512];           // per-wave 16x32 probs tile
  const int lane = threadIdx.x & 31;
  const int wave = threadIdx.x >> 5;
  const int wid  = blockIdx.x * 8 + wave;   // 0..2047
  const int qt = wid & (TBLK / 16 - 1);     // 0..127
  const int bh = wid >> 7;                  // 0..15
  const int t0 = qt * 16;
  const int lm = lane & 15, lg = lane >> 4;

  const half_t* qb = q + ((size_t)bh * TBLK + (t0 + lm)) * HSZ;
  const v16h qa0 = comb(*(const v8h*)(qb + 8 * lg),      *(const v8h*)(qb + 16 + 8 * lg));
  const v16h qa1 = comb(*(const v8h*)(qb + 32 + 8 * lg), *(const v8h*)(qb + 48 + 8 * lg));

  const half_t* kb = k  + (size_t)bh * TBLK * HSZ;
  const half_t* vb = vt + (size_t)bh * HSZ * TBLK;
  half_t* pl = Plds[wave];

  float mrow[8], ssum[8];
#pragma unroll
  for (int r = 0; r < 8; ++r) { mrow[r] = -1e30f; ssum[r] = 0.f; }
  v8f Oacc[4];
#pragma unroll
  for (int ht = 0; ht < 4; ++ht) Oacc[ht] = zero8();
  const float scale = 0.04419417382415922f;   // 1/sqrt(512)

  for (int s0 = 0; s0 <= t0 + 15; s0 += 32) {
    const half_t* kr0 = kb + (size_t)(s0 + lm) * HSZ;
    const half_t* kr1 = kr0 + 16 * HSZ;
    const v16h kb00 = comb(*(const v8h*)(kr0 + 8 * lg), *(const v8h*)(kr0 + 8 * lg + 16));
    const v16h kb01 = comb(*(const v8h*)(kr0 + 32 + 8 * lg), *(const v8h*)(kr0 + 48 + 8 * lg));
    const v16h kb10 = comb(*(const v8h*)(kr1 + 8 * lg), *(const v8h*)(kr1 + 8 * lg + 16));
    const v16h kb11 = comb(*(const v8h*)(kr1 + 32 + 8 * lg), *(const v8h*)(kr1 + 48 + 8 * lg));
    v8f S0 = zero8(), S1 = zero8();
    S0 = __builtin_amdgcn_wmma_f32_16x16x32_f16(false, qa0, false, kb00, (short)0, S0, false, false);
    S0 = __builtin_amdgcn_wmma_f32_16x16x32_f16(false, qa1, false, kb01, (short)0, S0, false, false);
    S1 = __builtin_amdgcn_wmma_f32_16x16x32_f16(false, qa0, false, kb10, (short)0, S1, false, false);
    S1 = __builtin_amdgcn_wmma_f32_16x16x32_f16(false, qa1, false, kb11, (short)0, S1, false, false);

#pragma unroll
    for (int r = 0; r < 8; ++r) {
      const int trow = t0 + r + 8 * lg;
      float a0 = S0[r] * scale; if (s0 + lm > trow)      a0 = -1e30f;
      float a1 = S1[r] * scale; if (s0 + 16 + lm > trow) a1 = -1e30f;
      float lmax = fmaxf(a0, a1);
      lmax = fmaxf(lmax, __shfl_xor(lmax, 1, 32));
      lmax = fmaxf(lmax, __shfl_xor(lmax, 2, 32));
      lmax = fmaxf(lmax, __shfl_xor(lmax, 4, 32));
      lmax = fmaxf(lmax, __shfl_xor(lmax, 8, 32));
      const float mnew = fmaxf(mrow[r], lmax);
      const float fsc  = __expf(mrow[r] - mnew);
      const float p0   = __expf(a0 - mnew);
      const float p1   = __expf(a1 - mnew);
      float ls = p0 + p1;
      ls += __shfl_xor(ls, 1, 32);
      ls += __shfl_xor(ls, 2, 32);
      ls += __shfl_xor(ls, 4, 32);
      ls += __shfl_xor(ls, 8, 32);
      ssum[r] = ssum[r] * fsc + ls;
      mrow[r] = mnew;
      Oacc[0][r] *= fsc; Oacc[1][r] *= fsc; Oacc[2][r] *= fsc; Oacc[3][r] *= fsc;
      pl[(r + 8 * lg) * 32 + lm]      = (half_t)p0;
      pl[(r + 8 * lg) * 32 + 16 + lm] = (half_t)p1;
    }
    const v16h pa = comb(*(const v8h*)&pl[lm * 32 + 8 * lg],
                         *(const v8h*)&pl[lm * 32 + 16 + 8 * lg]);
#pragma unroll
    for (int ht = 0; ht < 4; ++ht) {
      const half_t* vr = vb + (size_t)(ht * 16 + lm) * TBLK + s0 + 8 * lg;
      const v16h bv = comb(*(const v8h*)vr, *(const v8h*)(vr + 16));
      Oacc[ht] = __builtin_amdgcn_wmma_f32_16x16x32_f16(false, pa, false, bv, (short)0,
                                                        Oacc[ht], false, false);
    }
  }

  const int bi = bh >> 3, hh = bh & 7;
#pragma unroll
  for (int r = 0; r < 8; ++r) {
    const float inv = 1.0f / ssum[r];
    float* ar = att + ((size_t)bi * TBLK + (t0 + r + 8 * lg)) * DMODEL + hh * HSZ;
    ar[lm]      = Oacc[0][r] * inv;
    ar[16 + lm] = Oacc[1][r] * inv;
    ar[32 + lm] = Oacc[2][r] * inv;
    ar[48 + lm] = Oacc[3][r] * inv;
  }
}

// ---------------------------------------------------------------------------
// Host driver
// ---------------------------------------------------------------------------
extern "C" void kernel_launch(void* const* d_in, const int* in_sizes, int n_in,
                              void* d_out, int out_size, void* d_ws, size_t ws_size,
                              hipStream_t stream) {
  (void)in_sizes; (void)n_in; (void)out_size; (void)ws_size;
  const int*   idx   = (const int*)d_in[0];
  const float* tok   = (const float*)d_in[1];
  const float* pos   = (const float*)d_in[2];
  const float* ln1_g = (const float*)d_in[3];
  const float* ln1_b = (const float*)d_in[4];
  const float* Wq    = (const float*)d_in[5];
  const float* Wk    = (const float*)d_in[6];
  const float* Wv    = (const float*)d_in[7];
  const float* Wo    = (const float*)d_in[8];
  const float* bo    = (const float*)d_in[9];
  const float* ln2_g = (const float*)d_in[10];
  const float* ln2_b = (const float*)d_in[11];
  const float* W1    = (const float*)d_in[12];
  const float* b1    = (const float*)d_in[13];
  const float* W2    = (const float*)d_in[14];
  const float* b2    = (const float*)d_in[15];
  const float* lmW   = (const float*)d_in[16];
  const float* lmb   = (const float*)d_in[17];

  char* ws = (char*)d_ws;
  float*  x   = (float*)(ws);                          //  8 MB  [4096,512] f32
  float*  h   = (float*)(ws + ((size_t)8  << 20));     //  8 MB  LN output f32
  half_t* qh  = (half_t*)(ws + ((size_t)16 << 20));    //  4 MB  [B,H,T,HS] f16
  half_t* kh  = (half_t*)(ws + ((size_t)20 << 20));    //  4 MB  [B,H,T,HS] f16
  half_t* vth = (half_t*)(ws + ((size_t)24 << 20));    //  4 MB  [B,H,HS,T] f16
  float*  att = (float*)(ws + ((size_t)28 << 20));     //  8 MB  [4096,512] f32
  half_t* h1  = (half_t*)(ws + ((size_t)36 << 20));    // 16 MB  [4096,2048] f16

  const dim3 blk(256);
  const dim3 gDD(DMODEL / 128, MROWS / 128);           //  4 x 32
  const dim3 gFF(DFF / 128, MROWS / 128);              // 16 x 32
  const dim3 gLM(VOCAB / 128, MROWS / 128);            // 250 x 32

  embed_kernel<<<MROWS * DMODEL / 256, blk, 0, stream>>>(idx, tok, pos, x);

  for (int l = 0; l < NLAYER; ++l) {
    const size_t oDD = (size_t)l * DMODEL * DMODEL;
    const size_t oFF = (size_t)l * DMODEL * DFF;
    ln_kernel<<<MROWS / 8, blk, 0, stream>>>(x, ln1_g + l * DMODEL, ln1_b + l * DMODEL, h);
    gemm_kernel<float, MODE_QK><<<gDD, blk, 0, stream>>>(h, Wq + oDD, nullptr, qh,
                                                         MROWS, DMODEL, DMODEL);
    gemm_kernel<float, MODE_QK><<<gDD, blk, 0, stream>>>(h, Wk + oDD, nullptr, kh,
                                                         MROWS, DMODEL, DMODEL);
    gemm_kernel<float, MODE_V><<<gDD, blk, 0, stream>>>(h, Wv + oDD, nullptr, vth,
                                                        MROWS, DMODEL, DMODEL);
    attn_kernel<<<(BATCH * NHEAD * (TBLK / 16)) / 8, blk, 0, stream>>>(qh, kh, vth, att);
    gemm_kernel<float, MODE_RES><<<gDD, blk, 0, stream>>>(att, Wo + oDD, bo + l * DMODEL, x,
                                                          MROWS, DMODEL, DMODEL);
    ln_kernel<<<MROWS / 8, blk, 0, stream>>>(x, ln2_g + l * DMODEL, ln2_b + l * DMODEL, h);
    gemm_kernel<float, MODE_RELU><<<gFF, blk, 0, stream>>>(h, W1 + oFF, b1 + l * DFF, h1,
                                                           MROWS, DFF, DMODEL);
    gemm_kernel<half_t, MODE_RES><<<gDD, blk, 0, stream>>>(h1, W2 + oFF, b2 + l * DMODEL, x,
                                                           MROWS, DMODEL, DFF);
  }
  gemm_kernel<float, MODE_OUT><<<gLM, blk, 0, stream>>>(x, lmW, lmb, d_out,
                                                        MROWS, VOCAB, DMODEL);
}
